// AVAlign_31379031065132
// MI455X (gfx1250) — matrix operand order repaired and checked by reading, not compiled, not run on hardware
//
#include <hip/hip_runtime.h>
#include <hip/hip_bf16.h>

#define S_      32
#define FRAME_  8
#define CLS_    32
#define D_      512
#define HWA2    64      // 8*8
#define HWV2    196     // 14*14
#define DOUT_   128
#define BA_     1024    // S*CLS
#define BV_     256     // S*FRAME

typedef __attribute__((ext_vector_type(16))) __bf16 v16bf;
typedef __attribute__((ext_vector_type(8)))  __bf16 v8bf;
typedef __attribute__((ext_vector_type(4)))  __bf16 v4bf;
typedef __attribute__((ext_vector_type(2)))  __bf16 v2bf;
typedef __attribute__((ext_vector_type(8)))  float  v8f;

static __device__ __forceinline__ int lane_id() { return threadIdx.x & 31; }

// K-offset pattern for 16-bit A fragment (16x32), per ISA 7.12.2:
// VGPR v<4 : K = 2v + 8*half ; VGPR v>=4 : K = 16 + 2(v-4) + 8*half
static __device__ __forceinline__ int a_koff(int v, int h) {
  return (v < 4) ? (2 * v + 8 * h) : (16 + 2 * (v - 4) + 8 * h);
}

// ---- vectorized bf16 fragment loaders (two 16B chunks per lane)
static __device__ __forceinline__ v16bf frag_a_bf16v(const __bf16* src, int ld) {
  int l = lane_id(); int row = l & 15; int h = l >> 4;
  const __bf16* p = src + row * ld + 8 * h;
  v8bf lo = *(const v8bf*)(p);          // K in [8h, 8h+8)
  v8bf hi = *(const v8bf*)(p + 16);     // K in [16+8h, 16+8h+8)
  return __builtin_shufflevector(lo, hi, 0,1,2,3,4,5,6,7,8,9,10,11,12,13,14,15);
}

static __device__ __forceinline__ v16bf frag_b_bf16v(const __bf16* src, int ld) {
  int l = lane_id(); int n = l & 15; int h = l >> 4;
  const __bf16* p = src + n * ld + 8 * h;    // K in [8h, 8h+8) and [16+8h, 16+8h+8)
  v8bf lo = *(const v8bf*)(p);
  v8bf hi = *(const v8bf*)(p + 16);
  return __builtin_shufflevector(lo, hi, 0,1,2,3,4,5,6,7,8,9,10,11,12,13,14,15);
}

// ---- predicated f32 loaders (only for the K=196 cam/feat_v GEMM)
static __device__ __forceinline__ v16bf frag_a_f32_pred(const float* src, int ld,
                                                        int kb, int kmax) {
  int l = lane_id(); int row = l & 15; int h = l >> 4;
  v16bf a;
#pragma unroll
  for (int v = 0; v < 8; ++v) {
    int k = kb + a_koff(v, h);
    a[2 * v]     = (k     < kmax) ? (__bf16)src[row * ld + k]     : (__bf16)0.0f;
    a[2 * v + 1] = (k + 1 < kmax) ? (__bf16)src[row * ld + k + 1] : (__bf16)0.0f;
  }
  return a;
}

static __device__ __forceinline__ v16bf frag_b_f32_pred(const float* src, int ld,
                                                        int kb, int kmax) {
  int l = lane_id(); int n = l & 15; int h = l >> 4;
  v16bf b;
#pragma unroll
  for (int v = 0; v < 8; ++v) {
    int k = kb + a_koff(v, h);
    b[2 * v]     = (k     < kmax) ? (__bf16)src[n * ld + k]     : (__bf16)0.0f;
    b[2 * v + 1] = (k + 1 < kmax) ? (__bf16)src[n * ld + k + 1] : (__bf16)0.0f;
  }
  return b;
}

static __device__ __forceinline__ v8f wmma_bf16(v16bf a, v16bf b, v8f c) {
  return __builtin_amdgcn_wmma_f32_16x16x32_bf16(false, a, false, b,
                                                 (short)0, c, false, false);
}

static __device__ __forceinline__ v8f v8f_zero() {
  v8f z = {0.f, 0.f, 0.f, 0.f, 0.f, 0.f, 0.f, 0.f};
  return z;
}

// ---------------- Kernel 0: f32 -> bf16 conversion (weights)
__global__ __launch_bounds__(256) void k_cvt(const float* __restrict__ src,
                                             __bf16* __restrict__ dst, int n) {
  int i = (blockIdx.x * 256 + threadIdx.x) * 4;
  if (i + 3 < n) {
    float4 f = *(const float4*)(src + i);
    v4bf p = {(__bf16)f.x, (__bf16)f.y, (__bf16)f.z, (__bf16)f.w};
    *(v4bf*)(dst + i) = p;
  }
}

// ---------------- Kernel 1: fa[b,o] = max_hw( feat_a[b,:,hw] . Wt[o,:] ) + bt[o]
// grid (1024, 2), block 256. Wave w: o tiles {o0, o0+16}, o0 = by*256 + w*32,
// all 64 hw rows (4 m-tiles). A staged per-K-chunk in LDS as bf16 (ld 40).
__global__ __launch_bounds__(256) void k_fa(const float* __restrict__ feat_a,
                                            const __bf16* __restrict__ Wt_bf,
                                            const float* __restrict__ bt,
                                            __bf16* __restrict__ fa_bf) {
  __shared__ __bf16 As[HWA2 * 40];              // [hw][k], ld 40 (80B rows)
  int b  = blockIdx.x;
  int o0 = blockIdx.y * 256 + (threadIdx.x >> 5) * 32;
  const float* src = feat_a + (size_t)b * D_ * HWA2;   // [i][hw]
  int hw  = threadIdx.x & 63;
  int kp0 = threadIdx.x >> 6;                   // 0..3
  v8f acc[8];
#pragma unroll
  for (int i = 0; i < 8; ++i) acc[i] = v8f_zero();

  for (int kb = 0; kb < D_; kb += 32) {
    __syncthreads();
#pragma unroll
    for (int j = 0; j < 4; ++j) {               // convert-once staging
      int k = (j * 4 + kp0) * 2;
      float f0 = src[(size_t)(kb + k)     * HWA2 + hw];
      float f1 = src[(size_t)(kb + k + 1) * HWA2 + hw];
      v2bf p = {(__bf16)f0, (__bf16)f1};
      *(v2bf*)&As[hw * 40 + k] = p;
    }
    if (kb + 32 < D_)                           // hint next chunk toward caches
      __builtin_prefetch(src + (size_t)(kb + 32 + kp0 * 8) * HWA2 + hw, 0, 0);
    __syncthreads();
    v16bf bf0 = frag_b_bf16v(Wt_bf + (size_t)o0 * D_ + kb, D_);
    v16bf bf1 = frag_b_bf16v(Wt_bf + (size_t)(o0 + 16) * D_ + kb, D_);
#pragma unroll
    for (int mt = 0; mt < 4; ++mt) {
      v16bf af = frag_a_bf16v(&As[mt * 16 * 40], 40);
      acc[mt]     = wmma_bf16(af, bf0, acc[mt]);
      acc[4 + mt] = wmma_bf16(af, bf1, acc[4 + mt]);
    }
  }
  int l = lane_id(); int n = l & 15;
#pragma unroll
  for (int nt = 0; nt < 2; ++nt) {
    float m = -3.402823466e+38f;
#pragma unroll
    for (int mt = 0; mt < 4; ++mt)
#pragma unroll
      for (int v = 0; v < 8; ++v) m = fmaxf(m, acc[nt * 4 + mt][v]);
    m = fmaxf(m, __shfl_xor(m, 16, 32));
    if (l < 16) {
      int o = o0 + nt * 16 + n;
      fa_bf[(size_t)b * D_ + o] = (__bf16)(m + bt[o]);
    }
  }
}

// ---------------- Kernel 2: camsum[b*32+c] = sum_hw cam[b,c,hw]
__global__ __launch_bounds__(256) void k_camsum(const float* __restrict__ cam,
                                                float* __restrict__ camsum) {
  int w = blockIdx.x * 8 + (threadIdx.x >> 5);  // 0..255
  const float* row = cam + (size_t)w * HWV2;
  float s = 0.f;
  for (int k = lane_id(); k < HWV2; k += 32) s += row[k];
#pragma unroll
  for (int off = 16; off; off >>= 1) s += __shfl_xor(s, off, 32);
  if (lane_id() == 0) camsum[w] = s;
}

// ---------------- Kernel 3: g[b,c,i] = sum_hw cam[b,c,hw] * feat_v[b,i,hw] (bf16 out)
// per-b GEMM: A = cam (32 x 196), B = feat_v^T (196 x 512), K padded to 224.
__global__ __launch_bounds__(256) void k_g(const float* __restrict__ cam,
                                           const float* __restrict__ feat_v,
                                           __bf16* __restrict__ g_bf) {
  int b  = blockIdx.x;
  int i0 = blockIdx.y * 128 + (threadIdx.x >> 5) * 16;
  const float* arow = cam    + (size_t)b * CLS_ * HWV2;                    // [c][196]
  const float* brow = feat_v + (size_t)b * D_ * HWV2 + (size_t)i0 * HWV2;  // [i][196]
  v8f acc[2] = {v8f_zero(), v8f_zero()};
  for (int kb = 0; kb < 224; kb += 32) {
    v16bf bf = frag_b_f32_pred(brow, HWV2, kb, HWV2);
#pragma unroll
    for (int mt = 0; mt < 2; ++mt) {
      v16bf af = frag_a_f32_pred(arow + (size_t)(mt * 16) * HWV2, HWV2, kb, HWV2);
      acc[mt] = wmma_bf16(af, bf, acc[mt]);
    }
  }
  int l = lane_id(); int n = l & 15; int h = l >> 4;
#pragma unroll
  for (int mt = 0; mt < 2; ++mt)
#pragma unroll
    for (int v = 0; v < 8; ++v) {
      int c = mt * 16 + v + h * 8;
      g_bf[((size_t)b * CLS_ + c) * D_ + i0 + n] = (__bf16)acc[mt][v];
    }
}

// ---------------- Kernel 4 (fused): fg = norm(g @ Ws^T + bs*camsum); tv = fg @ Wv^T + bv
// grid 256 (32 rows each), block 256. GEMM1 writes bf16 tile to LDS; GEMM2 reads it.
__global__ __launch_bounds__(256) void k_fg_tv(const __bf16* __restrict__ g_bf,
                                               const float* __restrict__ camsum,
                                               const __bf16* __restrict__ Ws_bf,
                                               const float* __restrict__ bs,
                                               const __bf16* __restrict__ Wv_bf,
                                               const float* __restrict__ bv,
                                               float* __restrict__ tv) {
  __shared__ __bf16 fgl[32 * 520];              // [local row][d], ld 520 (1040B rows)
  int rbase = blockIdx.x * 32;                  // row = b*32 + c
  int wave  = threadIdx.x >> 5;
  int l = lane_id(); int n = l & 15; int h = l >> 4;

  // GEMM1: 32 rows x 512 d, each wave covers 16 d per phase, 4 phases
  for (int p = 0; p < 4; ++p) {
    int d0 = p * 128 + wave * 16;
    v8f acc[2] = {v8f_zero(), v8f_zero()};
    for (int kb = 0; kb < D_; kb += 32) {
      v16bf bf = frag_b_bf16v(Ws_bf + (size_t)d0 * D_ + kb, D_);
#pragma unroll
      for (int mt = 0; mt < 2; ++mt) {
        v16bf af = frag_a_bf16v(g_bf + (size_t)(rbase + mt * 16) * D_ + kb, D_);
        acc[mt] = wmma_bf16(af, bf, acc[mt]);
      }
    }
    int d = d0 + n;
    float bsd = bs[d];
#pragma unroll
    for (int mt = 0; mt < 2; ++mt)
#pragma unroll
      for (int v = 0; v < 8; ++v) {
        int lr = mt * 16 + v + h * 8;
        float cs = camsum[rbase + lr];
        float val = (acc[mt][v] + bsd * cs) / (cs + 1e-10f);
        fgl[lr * 520 + d] = (__bf16)val;
      }
  }
  __syncthreads();

  // GEMM2: 32 rows x 128 o, each wave one 16-o tile, K=512 from LDS
  int o0 = wave * 16;
  v8f acc2[2] = {v8f_zero(), v8f_zero()};
  for (int kb = 0; kb < D_; kb += 32) {
    v16bf bf = frag_b_bf16v(Wv_bf + (size_t)o0 * D_ + kb, D_);
#pragma unroll
    for (int mt = 0; mt < 2; ++mt) {
      v16bf af = frag_a_bf16v(&fgl[(mt * 16) * 520 + kb], 520);
      acc2[mt] = wmma_bf16(af, bf, acc2[mt]);
    }
  }
  int o = o0 + n;
  float bvo = bv[o];
#pragma unroll
  for (int mt = 0; mt < 2; ++mt)
#pragma unroll
    for (int v = 0; v < 8; ++v) {
      int r = rbase + mt * 16 + v + h * 8;
      tv[(size_t)r * DOUT_ + o] = acc2[mt][v] + bvo;
    }
}

// ---------------- Kernel 5: ta = fa @ Wa^T + ba  (1024 x 128, K=512)
__global__ __launch_bounds__(256) void k_ta(const __bf16* __restrict__ fa_bf,
                                            const __bf16* __restrict__ Wa_bf,
                                            const float* __restrict__ ba,
                                            float* __restrict__ ta) {
  int rbase = blockIdx.x * 64;
  int o0 = (threadIdx.x >> 5) * 16;
  v8f acc[4];
#pragma unroll
  for (int mt = 0; mt < 4; ++mt) acc[mt] = v8f_zero();
  for (int kb = 0; kb < D_; kb += 32) {
    v16bf bf = frag_b_bf16v(Wa_bf + (size_t)o0 * D_ + kb, D_);
#pragma unroll
    for (int mt = 0; mt < 4; ++mt) {
      v16bf af = frag_a_bf16v(fa_bf + (size_t)(rbase + mt * 16) * D_ + kb, D_);
      acc[mt] = wmma_bf16(af, bf, acc[mt]);
    }
  }
  int l = lane_id(); int n = l & 15; int h = l >> 4;
  int o = o0 + n;
  float bao = ba[o];
#pragma unroll
  for (int mt = 0; mt < 4; ++mt)
#pragma unroll
    for (int v = 0; v < 8; ++v) {
      int r = rbase + mt * 16 + v + h * 8;
      ta[(size_t)r * DOUT_ + o] = acc[mt][v] + bao;
    }
}

// ---------------- Kernel 6: losses. One wave per (s,c,f).
__global__ __launch_bounds__(256) void k_loss(const float* __restrict__ ta,
                                              const float* __restrict__ tv,
                                              const float* __restrict__ pred_a,
                                              const float* __restrict__ pred_v,
                                              const int* __restrict__ rand_frames,
                                              const int* __restrict__ rand_classes,
                                              float* __restrict__ out) {
  int w = blockIdx.x * 8 + (threadIdx.x >> 5);  // 0..8191
  int s = w >> 8; int rem = w & 255; int c = rem >> 3; int f = rem & 7;
  int l = lane_id();
  const float* tarow = ta + (size_t)(s * CLS_ + c) * DOUT_;
  const float* tvrow = tv + (size_t)((s * FRAME_ + f) * CLS_ + c) * DOUT_;
  int s2 = s ^ 1;
  int ridx = (s * CLS_ + c) * FRAME_ + f;
  int rf = rand_frames[ridx];
  int rc = rand_classes[ridx];
  const float* tdrow = tv + (size_t)((s2 * FRAME_ + rf) * CLS_ + rc) * DOUT_;
  float sco = 0.f, sdi = 0.f;
#pragma unroll
  for (int j = 0; j < 4; ++j) {
    int o = l * 4 + j;
    float t  = tarow[o];
    float d1 = t - tvrow[o]; sco += d1 * d1;
    float d2 = t - tdrow[o]; sdi += d2 * d2;
  }
#pragma unroll
  for (int off = 16; off; off >>= 1) {
    sco += __shfl_xor(sco, off, 32);
    sdi += __shfl_xor(sdi, off, 32);
  }
  if (l == 0) {
    float pa = pred_a[s * CLS_ + c];
    bool aa = pa > 0.3f;
    float pvv = pred_v[(s * FRAME_ + f) * CLS_ + c];
    bool av = (1.f / (1.f + __expf(-pvv))) > 0.3f;
    int num = (int)(pa * (float)FRAME_);
    float mco = (aa && av) ? 1.f : 0.f;
    float mdi = (aa && (f < num)) ? 1.f : 0.f;
    out[ridx] = sco * (1.f / (float)DOUT_) * mco;
    out[S_ * CLS_ * FRAME_ + ridx] = sdi * (1.f / (float)DOUT_) * mdi;
  }
}

extern "C" void kernel_launch(void* const* d_in, const int* in_sizes, int n_in,
                              void* d_out, int out_size, void* d_ws, size_t ws_size,
                              hipStream_t stream) {
  const float* feat_a = (const float*)d_in[0];
  const float* pred_a = (const float*)d_in[1];
  const float* feat_v = (const float*)d_in[2];
  const float* pred_v = (const float*)d_in[3];
  const float* cam    = (const float*)d_in[4];
  const int*   rand_frames  = (const int*)d_in[5];
  const int*   rand_classes = (const int*)d_in[6];
  const float* Wt = (const float*)d_in[7];
  const float* bt = (const float*)d_in[8];
  const float* Ws = (const float*)d_in[9];
  const float* bs = (const float*)d_in[10];
  const float* Wa = (const float*)d_in[11];
  const float* ba = (const float*)d_in[12];
  const float* Wv = (const float*)d_in[13];
  const float* bv = (const float*)d_in[14];

  char* ws = (char*)d_ws;
  const size_t MB = 1024 * 1024;
  __bf16* fa_bf  = (__bf16*)(ws);                      // 1024*512*2 = 1 MB
  __bf16* g_bf   = (__bf16*)(ws + 1 * MB);             // 8192*512*2 = 8 MB
  float*  camsum = (float*) (ws + 9 * MB);             // 8192*4
  float*  tv     = (float*) (ws + 9 * MB + 65536);     // 8192*128*4 = 4 MB
  float*  ta     = (float*) (ws + 14 * MB);            // 1024*128*4 = 0.5 MB
  __bf16* Wt_bf  = (__bf16*)(ws + 15 * MB);            // 512 KB
  __bf16* Ws_bf  = (__bf16*)(ws + 15 * MB + 524288);   // 512 KB
  __bf16* Wa_bf  = (__bf16*)(ws + 16 * MB);            // 128 KB
  __bf16* Wv_bf  = (__bf16*)(ws + 16 * MB + 131072);   // 128 KB
  float* out = (float*)d_out;

  k_cvt   <<<dim3(256),    256, 0, stream>>>(Wt, Wt_bf, D_ * D_);
  k_cvt   <<<dim3(256),    256, 0, stream>>>(Ws, Ws_bf, D_ * D_);
  k_cvt   <<<dim3(64),     256, 0, stream>>>(Wa, Wa_bf, DOUT_ * D_);
  k_cvt   <<<dim3(64),     256, 0, stream>>>(Wv, Wv_bf, DOUT_ * D_);
  k_fa    <<<dim3(BA_, 2), 256, 0, stream>>>(feat_a, Wt_bf, bt, fa_bf);
  k_camsum<<<dim3(32),     256, 0, stream>>>(cam, camsum);
  k_g     <<<dim3(BV_, 4), 256, 0, stream>>>(cam, feat_v, g_bf);
  k_fg_tv <<<dim3(256),    256, 0, stream>>>(g_bf, camsum, Ws_bf, bs, Wv_bf, bv, tv);
  k_ta    <<<dim3(16),     256, 0, stream>>>(fa_bf, Wa_bf, ba, ta);
  k_loss  <<<dim3(1024),   256, 0, stream>>>(ta, tv, pred_a, pred_v,
                                             rand_frames, rand_classes, out);
}
